// Llama3Model_65987877536217
// MI455X (gfx1250) — matrix operand order repaired and checked by reading, not compiled, not run on hardware
//
#include <hip/hip_runtime.h>
#include <math.h>

// ---------------------------------------------------------------------------
// Llama3-style forward for MI455X (gfx1250, wave32, WMMA).
// All matmuls on v_wmma_f32_16x16x32_bf16 (f32 accum). Operands are converted
// to bf16 by the producing kernel (rmsnorm / silu / attention epilogue) or a
// tiny weight-convert pre-pass, so the GEMM K-loop moves pure bf16:
//   global --(GLOBAL_LOAD_ASYNC_TO_LDS_B128, ASYNCcnt)--> LDS (double-buffered)
//   LDS --ds_load_b128--> fragments --> 16x WMMA per 64-K stage per wave.
// Falls back to load+ds_store staging when the async builtin is unavailable.
// ---------------------------------------------------------------------------

typedef __bf16 bf16_t;
typedef __attribute__((ext_vector_type(16))) __bf16 v16bf;
typedef __attribute__((ext_vector_type(8)))  __bf16 v8bf;
typedef __attribute__((ext_vector_type(8)))  float  v8f;
typedef __attribute__((ext_vector_type(4)))  int    v4i;

#define NLAYERS 4
#define DM      2048
#define NH      16
#define NKV     4
#define HD      128
#define FF      5632
#define NV      32000
#define SQ      2048
#define RMS_EPS 1e-5f
#define ROPE_THETA 500000.0f

#if __has_builtin(__builtin_amdgcn_global_load_async_to_lds_b128)
#define ASYNC_LDS 1
#endif

static __device__ __forceinline__ v8f v8f_zero() {
  v8f z = {0.f, 0.f, 0.f, 0.f, 0.f, 0.f, 0.f, 0.f};
  return z;
}

// 16-byte global->LDS copy: async (ASYNCcnt) when available, else sync.
static __device__ __forceinline__ void copy16_g2l(const bf16_t* __restrict__ g,
                                                  bf16_t* __restrict__ l) {
#if defined(ASYNC_LDS)
  __builtin_amdgcn_global_load_async_to_lds_b128(
      (__attribute__((address_space(1))) v4i*)g,
      (__attribute__((address_space(3))) v4i*)l, 0, 0);
#else
  *(v8bf*)l = *(const v8bf*)g;
#endif
}

static __device__ __forceinline__ void wait_async_all() {
#if defined(ASYNC_LDS)
#if __has_builtin(__builtin_amdgcn_s_wait_asynccnt)
  __builtin_amdgcn_s_wait_asynccnt(0);
#else
  asm volatile("s_wait_asynccnt 0x0" ::: "memory");
#endif
#endif
}

// ---------------------------------------------------------------------------
// GEMM: C[M x N] = A[M x K] * W[N x K]^T (+ optional f32 residual)
// A, W bf16 row-major. Block tile 128(M) x 128(N) x 64(K), double-buffered
// LDS (64 KB), 256 threads = 8 waves. Wave (mt, ng) owns a 32x64 slab:
// per 64-K stage it loads 2 A-frags + 2x4 B-frags and fires 16 WMMAs, with
// B fragments reused across both M-subtiles.
// ---------------------------------------------------------------------------
__global__ __launch_bounds__(256) void gemm_wmma_bf16(
    const bf16_t* __restrict__ A, int lda,
    const bf16_t* __restrict__ W, int ldw,
    float* __restrict__ C, int ldc,
    const float* __restrict__ resid,
    int K)
{
  __shared__ __align__(32) bf16_t As[2][128 * 64];   // [m][k]
  __shared__ __align__(32) bf16_t Bs[2][128 * 64];   // [n][k]

  const int tid  = threadIdx.x;
  const int lane = tid & 31;
  const int wv   = tid >> 5;
  const int hl   = lane >> 4;
  const int l16  = lane & 15;
  const int m0   = blockIdx.y * 128;
  const int n0   = blockIdx.x * 128;
  const int mt   = wv >> 1;          // 0..3 : 32-row slab
  const int ng   = wv & 1;           // 0..1 : group of 4 N-subtiles

  v8f acc[2][4];
#pragma unroll
  for (int i = 0; i < 2; ++i)
#pragma unroll
    for (int j = 0; j < 4; ++j) acc[i][j] = v8f_zero();

  // Staging: 16B chunks; As and Bs are each 1024 chunks -> 4 per thread.
  auto stage = [&](int buf, int k0) {
#pragma unroll
    for (int j = 0; j < 4; ++j) {
      const int c = tid + j * 256;
      const int r = c >> 3, col = (c & 7) * 8;
      copy16_g2l(A + (size_t)(m0 + r) * lda + k0 + col, &As[buf][r * 64 + col]);
    }
#pragma unroll
    for (int j = 0; j < 4; ++j) {
      const int c = tid + j * 256;
      const int r = c >> 3, col = (c & 7) * 8;
      copy16_g2l(W + (size_t)(n0 + r) * ldw + k0 + col, &Bs[buf][r * 64 + col]);
    }
  };

  const int nk = K >> 6;
  stage(0, 0);
  for (int ks = 0; ks < nk; ++ks) {
    const int cur = ks & 1;
    wait_async_all();                 // staging of `cur` has landed
    __syncthreads();                  // visible to all waves; prev reads done
    if (ks + 1 < nk) stage(cur ^ 1, (ks + 1) * 64);  // overlaps with compute

#pragma unroll
    for (int kc = 0; kc < 2; ++kc) {
      const int kb = kc * 32;
      // Two A fragments (rows mt*32 + {0,16} + l16), A-frag K layout:
      // lane half hl -> K = hl*8+[0..7] and 16+hl*8+[0..7]
      v16bf af[2];
#pragma unroll
      for (int i = 0; i < 2; ++i) {
        const int arow = (mt * 32 + i * 16 + l16) * 64;
        v8bf alo = *(const v8bf*)&As[cur][arow + kb + hl * 8];
        v8bf ahi = *(const v8bf*)&As[cur][arow + kb + 16 + hl * 8];
        af[i] = __builtin_shufflevector(alo, ahi,
            0, 1, 2, 3, 4, 5, 6, 7, 8, 9, 10, 11, 12, 13, 14, 15);
      }
      // Four B fragments (cols (ng*4+j)*16 + l16), K = hl*8+[0..7] and 16+hl*8+[0..7].
      v16bf bfg[4];
#pragma unroll
      for (int j = 0; j < 4; ++j) {
        const int brow = ((ng * 4 + j) * 16 + l16) * 64;
        v8bf blo = *(const v8bf*)&Bs[cur][brow + kb + hl * 8];
        v8bf bhi = *(const v8bf*)&Bs[cur][brow + kb + 16 + hl * 8];
        bfg[j] = __builtin_shufflevector(blo, bhi,
            0, 1, 2, 3, 4, 5, 6, 7, 8, 9, 10, 11, 12, 13, 14, 15);
      }
      // Dense 8-WMMA burst; B frags reused across the two M subtiles.
#pragma unroll
      for (int i = 0; i < 2; ++i)
#pragma unroll
        for (int j = 0; j < 4; ++j)
          acc[i][j] = __builtin_amdgcn_wmma_f32_16x16x32_bf16(
              false, af[i], false, bfg[j], (short)0, acc[i][j], false, false);
    }
  }

#pragma unroll
  for (int i = 0; i < 2; ++i) {
#pragma unroll
    for (int j = 0; j < 4; ++j) {
      const int col = n0 + (ng * 4 + j) * 16 + l16;
#pragma unroll
      for (int r = 0; r < 8; ++r) {
        const int row = m0 + mt * 32 + i * 16 + hl * 8 + r;  // C: row=half*8+r
        float vvv = acc[i][j][r];
        if (resid) vvv += resid[(size_t)row * ldc + col];
        C[(size_t)row * ldc + col] = vvv;
      }
    }
  }
}

// ---------------------------------------------------------------------------
// Flash attention: block = 4 waves, grid = (SQ/64, NH). Each wave owns 16
// queries of one head. Q/K/V are f32 (post-RoPE); converted to bf16 at
// register/LDS staging. Output written directly as bf16 for the wo GEMM.
// ---------------------------------------------------------------------------
__global__ __launch_bounds__(128) void attention_kernel(
    const float* __restrict__ Q, const float* __restrict__ Kg,
    const float* __restrict__ Vg, bf16_t* __restrict__ Ob)
{
  __shared__ __align__(32) bf16_t Kt[32 * HD];     // [key][hd]
  __shared__ __align__(32) bf16_t Vt[HD * 32];     // [hd][key] (transposed)
  __shared__ __align__(32) bf16_t Pw[4][16 * 32];  // per-wave P tile

  const int tid  = threadIdx.x;
  const int lane = tid & 31;
  const int wv   = tid >> 5;
  const int hl   = lane >> 4;
  const int l16  = lane & 15;
  const int h    = blockIdx.y;
  const int kvh  = h >> 2;                          // H/KV = 4
  const int qb   = blockIdx.x * 64 + wv * 16;
  const float scale = 0.08838834764831845f;         // 1/sqrt(128) folded in Q

  v16bf qf[4];
  {
    const float* qp = Q + (size_t)(qb + l16) * (NH * HD) + (size_t)h * HD;
#pragma unroll
    for (int c = 0; c < 4; ++c) {
      v16bf a;
#pragma unroll
      for (int j = 0; j < 8; ++j) {
        a[j]     = (bf16_t)(qp[c * 32 + hl * 8 + j] * scale);
        a[8 + j] = (bf16_t)(qp[c * 32 + 16 + hl * 8 + j] * scale);
      }
      qf[c] = a;
    }
  }

  v8f o[8];
#pragma unroll
  for (int nt = 0; nt < 8; ++nt) o[nt] = v8f_zero();
  float rm[8], rl[8];
#pragma unroll
  for (int r = 0; r < 8; ++r) { rm[r] = -1e30f; rl[r] = 0.f; }

  const int nkt = (blockIdx.x + 1) * 2;
  for (int kt = 0; kt < nkt; ++kt) {
    const int kb = kt * 32;
    __syncthreads();
    {   // stage 32 keys x 128 hd of K and V (V transposed), bf16
      const int key = tid >> 2;
      const int hd0 = (tid & 3) * 32;
      const float* kp = Kg + (size_t)(kb + key) * (NKV * HD) + (size_t)kvh * HD + hd0;
      const float* vp = Vg + (size_t)(kb + key) * (NKV * HD) + (size_t)kvh * HD + hd0;
#pragma unroll
      for (int j = 0; j < 32; ++j) {
        Kt[key * HD + hd0 + j]   = (bf16_t)kp[j];
        Vt[(hd0 + j) * 32 + key] = (bf16_t)vp[j];
      }
    }
    __syncthreads();

    if (kb <= qb + 15) {   // skip fully-masked tiles (barriers stay uniform)
      // Preload all K fragments, then a dense 8-WMMA burst.
      v16bf kf[8];
#pragma unroll
      for (int c = 0; c < 4; ++c) {
        v8bf k0l = *(const v8bf*)&Kt[l16 * HD + c * 32 + hl * 8];
        v8bf k0h = *(const v8bf*)&Kt[l16 * HD + c * 32 + 16 + hl * 8];
        v8bf k1l = *(const v8bf*)&Kt[(16 + l16) * HD + c * 32 + hl * 8];
        v8bf k1h = *(const v8bf*)&Kt[(16 + l16) * HD + c * 32 + 16 + hl * 8];
        kf[2 * c]     = __builtin_shufflevector(k0l, k0h, 0, 1, 2, 3, 4, 5, 6, 7, 8, 9, 10, 11, 12, 13, 14, 15);
        kf[2 * c + 1] = __builtin_shufflevector(k1l, k1h, 0, 1, 2, 3, 4, 5, 6, 7, 8, 9, 10, 11, 12, 13, 14, 15);
      }
      v8f s0 = v8f_zero(), s1 = v8f_zero();
#pragma unroll
      for (int c = 0; c < 4; ++c) {
        s0 = __builtin_amdgcn_wmma_f32_16x16x32_bf16(false, qf[c], false, kf[2 * c],     (short)0, s0, false, false);
        s1 = __builtin_amdgcn_wmma_f32_16x16x32_bf16(false, qf[c], false, kf[2 * c + 1], (short)0, s1, false, false);
      }
      // causal mask + online softmax (row r of this half <-> lanes of half)
#pragma unroll
      for (int r = 0; r < 8; ++r) {
        const int qrow = qb + hl * 8 + r;
        float e0 = (kb + l16      <= qrow) ? s0[r] : -1e30f;
        float e1 = (kb + 16 + l16 <= qrow) ? s1[r] : -1e30f;
        float mx = fmaxf(e0, e1);
#pragma unroll
        for (int off = 1; off < 16; off <<= 1) mx = fmaxf(mx, __shfl_xor(mx, off, 32));
        const float mn = fmaxf(rm[r], mx);
        const float cf = __expf(rm[r] - mn);
        const float p0 = __expf(e0 - mn);
        const float p1 = __expf(e1 - mn);
        float ps = p0 + p1;
#pragma unroll
        for (int off = 1; off < 16; off <<= 1) ps += __shfl_xor(ps, off, 32);
        rl[r] = rl[r] * cf + ps;
        rm[r] = mn;
#pragma unroll
        for (int nt = 0; nt < 8; ++nt) o[nt][r] *= cf;
        const int m = hl * 8 + r;
        Pw[wv][m * 32 + l16]      = (bf16_t)p0;
        Pw[wv][m * 32 + 16 + l16] = (bf16_t)p1;
      }
      // per-wave LDS round-trip: drain our ds stores before re-reading
      asm volatile("s_wait_dscnt 0x0" ::: "memory");
      v8bf plo = *(const v8bf*)&Pw[wv][l16 * 32 + hl * 8];
      v8bf phi = *(const v8bf*)&Pw[wv][l16 * 32 + 16 + hl * 8];
      v16bf pf = __builtin_shufflevector(plo, phi,
          0, 1, 2, 3, 4, 5, 6, 7, 8, 9, 10, 11, 12, 13, 14, 15);
      v16bf vf[8];
#pragma unroll
      for (int nt = 0; nt < 8; ++nt) {
        v8bf vlo = *(const v8bf*)&Vt[(nt * 16 + l16) * 32 + hl * 8];
        v8bf vhi = *(const v8bf*)&Vt[(nt * 16 + l16) * 32 + 16 + hl * 8];
        vf[nt] = __builtin_shufflevector(vlo, vhi, 0, 1, 2, 3, 4, 5, 6, 7, 8, 9, 10, 11, 12, 13, 14, 15);
      }
#pragma unroll
      for (int nt = 0; nt < 8; ++nt)
        o[nt] = __builtin_amdgcn_wmma_f32_16x16x32_bf16(
            false, pf, false, vf[nt], (short)0, o[nt], false, false);
    }
  }

#pragma unroll
  for (int nt = 0; nt < 8; ++nt) {
#pragma unroll
    for (int r = 0; r < 8; ++r) {
      const int row = qb + hl * 8 + r;
      Ob[(size_t)row * (NH * HD) + (size_t)h * HD + nt * 16 + l16] =
          (bf16_t)(o[nt][r] / rl[r]);
    }
  }
}

// ---------------------------------------------------------------------------
// Elementwise / reduction kernels
// ---------------------------------------------------------------------------
static __device__ __forceinline__ float block_sum256(float v) {
  __shared__ float red[8];
#pragma unroll
  for (int off = 16; off > 0; off >>= 1) v += __shfl_xor(v, off, 32);
  if ((threadIdx.x & 31) == 0) red[threadIdx.x >> 5] = v;
  __syncthreads();
  float t = 0.f;
#pragma unroll
  for (int i = 0; i < 8; ++i) t += red[i];
  __syncthreads();
  return t;
}

// RMSNorm producing bf16 (feeds GEMM A operand)
__global__ __launch_bounds__(256) void rmsnorm_bf16_kernel(
    const float* __restrict__ x, const float* __restrict__ w,
    bf16_t* __restrict__ out)
{
  const int row = blockIdx.x;
  const float* xr = x + (size_t)row * DM;
  float ss = 0.f;
  for (int i = threadIdx.x; i < DM; i += 256) { float v = xr[i]; ss += v * v; }
  const float tot = block_sum256(ss);
  const float r = rsqrtf(tot / (float)DM + RMS_EPS);
  for (int i = threadIdx.x; i < DM; i += 256)
    out[(size_t)row * DM + i] = (bf16_t)(xr[i] * r * w[i]);
}

// RMSNorm producing f32 (final norm feeding the logits GEMV)
__global__ __launch_bounds__(256) void rmsnorm_f32_kernel(
    const float* __restrict__ x, const float* __restrict__ w,
    float* __restrict__ out)
{
  const float* xr = x;
  float ss = 0.f;
  for (int i = threadIdx.x; i < DM; i += 256) { float v = xr[i]; ss += v * v; }
  const float tot = block_sum256(ss);
  const float r = rsqrtf(tot / (float)DM + RMS_EPS);
  for (int i = threadIdx.x; i < DM; i += 256) out[i] = xr[i] * r * w[i];
}

// fp32 -> bf16 weight tiles (8 elems/thread, b128 in / b128 out)
__global__ __launch_bounds__(256) void f32_to_bf16_kernel(
    const float* __restrict__ src, bf16_t* __restrict__ dst)
{
  const size_t i = ((size_t)blockIdx.x * 256 + threadIdx.x) * 8;
  float4 a0 = *(const float4*)(src + i);
  float4 a1 = *(const float4*)(src + i + 4);
  v8bf t;
  t[0] = (bf16_t)a0.x; t[1] = (bf16_t)a0.y; t[2] = (bf16_t)a0.z; t[3] = (bf16_t)a0.w;
  t[4] = (bf16_t)a1.x; t[5] = (bf16_t)a1.y; t[6] = (bf16_t)a1.z; t[7] = (bf16_t)a1.w;
  *(v8bf*)(dst + i) = t;
}

__global__ __launch_bounds__(256) void embed_kernel(
    const int* __restrict__ tok, const float* __restrict__ emb,
    float* __restrict__ x)
{
  const int s = blockIdx.x;
  const size_t base = (size_t)tok[s] * DM;
  for (int i = threadIdx.x; i < DM; i += 256)
    x[(size_t)s * DM + i] = emb[base + i];
}

__global__ __launch_bounds__(256) void rope_kernel(float* __restrict__ t, int nheads) {
  const int idx = blockIdx.x * 256 + threadIdx.x;   // (s, head, pair i)
  const int i  = idx & 63;
  const int hd = (idx >> 6) % nheads;
  const int s  = idx / (64 * nheads);
  const float freq = __expf(-(float)i * (logf(ROPE_THETA) / 64.f));
  const float ang  = (float)s * freq;
  float sn, cs;
  __sincosf(ang, &sn, &cs);
  float* p = t + ((size_t)s * nheads + hd) * HD + 2 * i;
  const float tr = p[0], ti = p[1];
  p[0] = tr * cs - ti * sn;
  p[1] = tr * sn + ti * cs;
}

// act = silu(g) * u, written as bf16 (feeds w2 GEMM A operand)
__global__ __launch_bounds__(256) void silu_mul_bf16_kernel(
    const float* __restrict__ g, const float* __restrict__ u,
    bf16_t* __restrict__ act)
{
  const size_t i = (size_t)blockIdx.x * 256 + threadIdx.x;
  const float gv = g[i];
  act[i] = (bf16_t)((gv / (1.f + __expf(-gv))) * u[i]);
}

// logits GEMV: wave per vocab row (memory-bound, 262MB ~ 11us @ 23.3TB/s)
__global__ __launch_bounds__(256) void logits_kernel(
    const float* __restrict__ xn, const float* __restrict__ wo,
    float* __restrict__ out)
{
  const int row  = blockIdx.x * 8 + (threadIdx.x >> 5);
  const int lane = threadIdx.x & 31;
  const float* wr = wo + (size_t)row * DM;
  float s = 0.f;
  for (int i = lane; i < DM; i += 32) s += xn[i] * wr[i];
#pragma unroll
  for (int off = 16; off > 0; off >>= 1) s += __shfl_xor(s, off, 32);
  if (lane == 0) out[row] = s;
}

// ---------------------------------------------------------------------------
extern "C" void kernel_launch(void* const* d_in, const int* in_sizes, int n_in,
                              void* d_out, int out_size, void* d_ws, size_t ws_size,
                              hipStream_t stream)
{
  (void)in_sizes; (void)n_in; (void)out_size; (void)ws_size;
  const int*   tokens    = (const int*)  d_in[0];
  const float* emb       = (const float*)d_in[1];
  const float* attn_norm = (const float*)d_in[2];
  const float* ffn_norm  = (const float*)d_in[3];
  const float* wq        = (const float*)d_in[4];
  const float* wk        = (const float*)d_in[5];
  const float* wvp       = (const float*)d_in[6];
  const float* wo        = (const float*)d_in[7];
  const float* w1        = (const float*)d_in[8];
  const float* w2        = (const float*)d_in[9];
  const float* w3        = (const float*)d_in[10];
  const float* final_nw  = (const float*)d_in[11];
  const float* out_w     = (const float*)d_in[12];
  float* logits = (float*)d_out;

  // Workspace layout (~195 MB), all regions 16B-aligned by construction.
  char* p = (char*)d_ws;
  float*  x    = (float*) p; p += (size_t)SQ * DM * 4;        // residual (f32)
  float*  q    = (float*) p; p += (size_t)SQ * DM * 4;        // q proj (f32)
  float*  kb   = (float*) p; p += (size_t)SQ * NKV * HD * 4;  // k proj (f32)
  float*  vb   = (float*) p; p += (size_t)SQ * NKV * HD * 4;  // v proj (f32)
  float*  gbuf = (float*) p; p += (size_t)SQ * FF * 4;        // w1 out (f32)
  float*  ubuf = (float*) p; p += (size_t)SQ * FF * 4;        // w3 out (f32)
  float*  xn   = (float*) p; p += (size_t)DM * 4;             // final norm row
  bf16_t* hb   = (bf16_t*)p; p += (size_t)SQ * DM * 2;        // norm out (bf16)
  bf16_t* obf  = (bf16_t*)p; p += (size_t)SQ * DM * 2;        // attn out (bf16)
  bf16_t* actb = (bf16_t*)p; p += (size_t)SQ * FF * 2;        // silu*u (bf16)
  bf16_t* wbuf = (bf16_t*)p; p += (size_t)FF * DM * 2;        // bf16 weight tile

  auto conv = [&](const float* src, size_t n) {
    f32_to_bf16_kernel<<<(unsigned)(n / 2048), 256, 0, stream>>>(src, wbuf);
  };

  embed_kernel<<<SQ, 256, 0, stream>>>(tokens, emb, x);

  for (int l = 0; l < NLAYERS; ++l) {
    const float* wql = wq  + (size_t)l * (NH * HD) * DM;
    const float* wkl = wk  + (size_t)l * (NKV * HD) * DM;
    const float* wvl = wvp + (size_t)l * (NKV * HD) * DM;
    const float* wol = wo  + (size_t)l * DM * (NH * HD);
    const float* w1l = w1  + (size_t)l * FF * DM;
    const float* w2l = w2  + (size_t)l * DM * FF;
    const float* w3l = w3  + (size_t)l * FF * DM;

    rmsnorm_bf16_kernel<<<SQ, 256, 0, stream>>>(x, attn_norm + (size_t)l * DM, hb);

    conv(wql, (size_t)(NH * HD) * DM);
    gemm_wmma_bf16<<<dim3((NH * HD) / 128, SQ / 128), 256, 0, stream>>>(
        hb, DM, wbuf, DM, q, NH * HD, nullptr, DM);
    conv(wkl, (size_t)(NKV * HD) * DM);
    gemm_wmma_bf16<<<dim3((NKV * HD) / 128, SQ / 128), 256, 0, stream>>>(
        hb, DM, wbuf, DM, kb, NKV * HD, nullptr, DM);
    conv(wvl, (size_t)(NKV * HD) * DM);
    gemm_wmma_bf16<<<dim3((NKV * HD) / 128, SQ / 128), 256, 0, stream>>>(
        hb, DM, wbuf, DM, vb, NKV * HD, nullptr, DM);

    rope_kernel<<<(SQ * NH * (HD / 2)) / 256, 256, 0, stream>>>(q, NH);
    rope_kernel<<<(SQ * NKV * (HD / 2)) / 256, 256, 0, stream>>>(kb, NKV);

    attention_kernel<<<dim3(SQ / 64, NH), 128, 0, stream>>>(q, kb, vb, obf);

    // x = x + O @ wo^T
    conv(wol, (size_t)DM * (NH * HD));
    gemm_wmma_bf16<<<dim3(DM / 128, SQ / 128), 256, 0, stream>>>(
        obf, NH * HD, wbuf, NH * HD, x, DM, x, NH * HD);

    rmsnorm_bf16_kernel<<<SQ, 256, 0, stream>>>(x, ffn_norm + (size_t)l * DM, hb);

    conv(w1l, (size_t)FF * DM);
    gemm_wmma_bf16<<<dim3(FF / 128, SQ / 128), 256, 0, stream>>>(
        hb, DM, wbuf, DM, gbuf, FF, nullptr, DM);
    conv(w3l, (size_t)FF * DM);
    gemm_wmma_bf16<<<dim3(FF / 128, SQ / 128), 256, 0, stream>>>(
        hb, DM, wbuf, DM, ubuf, FF, nullptr, DM);

    silu_mul_bf16_kernel<<<(unsigned)(((size_t)SQ * FF) / 256), 256, 0, stream>>>(
        gbuf, ubuf, actb);

    // x = x + act @ w2^T
    conv(w2l, (size_t)DM * FF);
    gemm_wmma_bf16<<<dim3(DM / 128, SQ / 128), 256, 0, stream>>>(
        actb, FF, wbuf, FF, x, DM, x, FF);
  }

  rmsnorm_f32_kernel<<<1, 256, 0, stream>>>(x + (size_t)(SQ - 1) * DM, final_nw, xn);
  logits_kernel<<<NV / 8, 256, 0, stream>>>(xn, out_w, logits);
}
